// ChannelRoutingGraphAttention_63943473103180
// MI455X (gfx1250) — matrix operand order repaired and checked by reading, not compiled, not run on hardware
//
#include <hip/hip_runtime.h>
#include <math.h>
#include <cstdint>

// ---------------------------------------------------------------------------
// ChannelRoutingGraphAttention for MI455X (gfx1250, wave32, WMMA)
// M=4096 nodes, N=32 neighbors, D=256, H=8 heads (Dh=32), RH=128
// Edge-level GEMMs collapsed to node-level WMMA GEMMs (tables L2-resident).
// Router = wave-per-edge kernel (no block barriers). Attention kernel is
// barrier-free after preload (wave == head, register softmax, shfl coefs).
// GEMM: f16-staged operands (A row-major, B transposed), 64x128 block tile,
// double-buffered LDS, all b128 traffic, K=256 fully unrolled (32 WMMAs).
// ---------------------------------------------------------------------------

#define MROWS 4096
#define NNBR  32
#define DDIM  256
#define NHEAD 8
#define RHID  128
#define KDIM  256                         // all GEMMs have K = 256
#define ATT_SCALE 0.17677669529663687f    // 1/sqrt(32)

typedef __attribute__((ext_vector_type(16))) _Float16 v16h;
typedef __attribute__((ext_vector_type(8)))  _Float16 v8h;
typedef __attribute__((ext_vector_type(8)))  float    v8f;

// ------------------------------ helpers ------------------------------------

__device__ __forceinline__ float wave_reduce_sum(float v) {
#pragma unroll
  for (int off = 16; off > 0; off >>= 1) v += __shfl_xor(v, off, 32);
  return v;
}

__device__ __forceinline__ float wave_reduce_max(float v) {
#pragma unroll
  for (int off = 16; off > 0; off >>= 1) v = fmaxf(v, __shfl_xor(v, off, 32));
  return v;
}

__device__ __forceinline__ float gelu_exact(float x) {
  return 0.5f * x * (1.0f + erff(x * 0.70710678118654752f));
}

__device__ __forceinline__ v16h make_v16(v8h lo, v8h hi) {
  v16h r;
#pragma unroll
  for (int i = 0; i < 8; ++i) { r[i] = lo[i]; r[i + 8] = hi[i]; }
  return r;
}

// --------------------------- staging kernels -------------------------------

__global__ __launch_bounds__(256)
void crga_cvt_f16(const float* __restrict__ in, _Float16* __restrict__ out, int n) {
  const int i = blockIdx.x * 256 + threadIdx.x;
  if (i < n) out[i] = (_Float16)in[i];
}

// B [K x N] fp32 row-major  ->  Bt [N x K] f16 row-major
__global__ __launch_bounds__(256)
void crga_transpose_f16(const float* __restrict__ B, _Float16* __restrict__ Bt,
                        int K, int N) {
  const int i = blockIdx.x * 256 + threadIdx.x;
  if (i >= K * N) return;
  const int k = i / N, n = i - k * N;
  Bt[(size_t)n * K + k] = (_Float16)B[i];
}

// ------------------------ WMMA f16 GEMM (f16 in, fp32 out) -----------------
// C[Mrows x Ncols] = A16[Mrows x 256] @ Bt16[Ncols x 256]^T (+bias) (+C)
// Block tile 64x128, 256 threads = 8 waves; wave -> 16x64 slab (4 WMMA/step).
// K fully unrolled (8 steps of 32), double-buffered LDS, all b128 traffic.

#define BLK_M 64
#define BLK_N 128
#define BLK_K 32
#define NSTEP (KDIM / BLK_K)
#define LDS_K (BLK_K + 8)   // f16 units; 80B row stride (16B aligned)

__global__ __launch_bounds__(256)
void crga_gemm_wmma(const _Float16* __restrict__ A16,
                    const _Float16* __restrict__ Bt16,
                    float* __restrict__ C, int ldc,
                    const float* __restrict__ bias,
                    _Float16* __restrict__ C16,
                    int addC)
{
  __shared__ __align__(16) _Float16 As[2][BLK_M][LDS_K];
  __shared__ __align__(16) _Float16 Bs[2][BLK_N][LDS_K];

  const int t    = threadIdx.x;
  const int wave = t >> 5;
  const int lane = t & 31;
  const int block_row = blockIdx.x * BLK_M;
  const int block_col = blockIdx.y * BLK_N;

  const int wr = wave & 3;                   // 16-row tile within 64
  const int wc = wave >> 2;                  // 64-col slab within 128
  const int mrow  = wr * 16 + (lane & 15);
  const int hh    = lane >> 4;               // half-wave
  const int nbase = wc * 64 + (lane & 15);

  // cooperative loader mapping: one v8h (16B) per slot
  const int a_row = t >> 2, a_kc = (t & 3) << 3;        // 64 rows x 4 chunks
  const _Float16* Aptr  = A16  + (size_t)(block_row + a_row) * KDIM + a_kc;
  const _Float16* Bptr0 = Bt16 + (size_t)(block_col + a_row) * KDIM + a_kc;
  const _Float16* Bptr1 = Bptr0 + (size_t)64 * KDIM;

  v8f acc0 = {}, acc1 = {}, acc2 = {}, acc3 = {};

  // preload step 0 into buffer 0
  {
    const v8h av  = *(const v8h*)(Aptr);
    const v8h bv0 = *(const v8h*)(Bptr0);
    const v8h bv1 = *(const v8h*)(Bptr1);
    *(v8h*)&As[0][a_row][a_kc]      = av;
    *(v8h*)&Bs[0][a_row][a_kc]      = bv0;
    *(v8h*)&Bs[0][a_row + 64][a_kc] = bv1;
  }
  __syncthreads();

  auto compute = [&](int cur) {
    const v16h af = make_v16(*(const v8h*)&As[cur][mrow][hh << 4],
                             *(const v8h*)&As[cur][mrow][(hh << 4) + 8]);
    const v16h b0 = make_v16(*(const v8h*)&Bs[cur][nbase][hh << 4],
                             *(const v8h*)&Bs[cur][nbase][(hh << 4) + 8]);
    const v16h b1 = make_v16(*(const v8h*)&Bs[cur][nbase + 16][hh << 4],
                             *(const v8h*)&Bs[cur][nbase + 16][(hh << 4) + 8]);
    const v16h b2 = make_v16(*(const v8h*)&Bs[cur][nbase + 32][hh << 4],
                             *(const v8h*)&Bs[cur][nbase + 32][(hh << 4) + 8]);
    const v16h b3 = make_v16(*(const v8h*)&Bs[cur][nbase + 48][hh << 4],
                             *(const v8h*)&Bs[cur][nbase + 48][(hh << 4) + 8]);
    acc0 = __builtin_amdgcn_wmma_f32_16x16x32_f16(false, af, false, b0, (short)0, acc0, false, false);
    acc1 = __builtin_amdgcn_wmma_f32_16x16x32_f16(false, af, false, b1, (short)0, acc1, false, false);
    acc2 = __builtin_amdgcn_wmma_f32_16x16x32_f16(false, af, false, b2, (short)0, acc2, false, false);
    acc3 = __builtin_amdgcn_wmma_f32_16x16x32_f16(false, af, false, b3, (short)0, acc3, false, false);
  };

#pragma unroll
  for (int step = 0; step < NSTEP; ++step) {
    const int cur = step & 1;
    if (step + 1 < NSTEP) {
      const int ko = (step + 1) * BLK_K;
      const v8h av  = *(const v8h*)(Aptr + ko);  // overlaps WMMAs below
      const v8h bv0 = *(const v8h*)(Bptr0 + ko);
      const v8h bv1 = *(const v8h*)(Bptr1 + ko);
      compute(cur);
      *(v8h*)&As[cur ^ 1][a_row][a_kc]      = av;
      *(v8h*)&Bs[cur ^ 1][a_row][a_kc]      = bv0;
      *(v8h*)&Bs[cur ^ 1][a_row + 64][a_kc] = bv1;
    } else {
      compute(cur);
    }
    __syncthreads();
  }

  // writeback: C/D layout n = lane&15 (per 16-col tile), m = v + 8*hh
  const int ncol0 = block_col + wc * 64 + (lane & 15);
  const int rbase = block_row + wr * 16 + (hh << 3);
#pragma unroll
  for (int cti = 0; cti < 4; ++cti) {
    const int col = ncol0 + cti * 16;
    const float bb = bias ? bias[col] : 0.0f;
    const v8f* accp = (cti == 0) ? &acc0 : (cti == 1) ? &acc1 : (cti == 2) ? &acc2 : &acc3;
#pragma unroll
    for (int v = 0; v < 8; ++v) {
      const int row = rbase + v;
      float r = (*accp)[v] + bb;
      if (addC) r += C[(size_t)row * ldc + col];
      C[(size_t)row * ldc + col] = r;
      if (C16) C16[(size_t)row * ldc + col] = (_Float16)r;
    }
  }
}

// --------------------- phase 2a: router (wave per edge) --------------------
// hid = gelu(RN[j] + RV[m]); logits = hid @ r_w2 + r_b2; pi = softmax (0 if
// masked). 8 waves/block, 4 RH elems per lane, shuffle reductions only.

__global__ __launch_bounds__(256)
void crga_router(const long long* __restrict__ nidx,
                 const unsigned char* __restrict__ nmask,
                 const float* __restrict__ RN, const float* __restrict__ RV,
                 const float* __restrict__ r_w2, const float* __restrict__ r_b2,
                 float* __restrict__ piTab)            // [M*N, 4]
{
  const int e    = blockIdx.x * 8 + (threadIdx.x >> 5);
  const int lane = threadIdx.x & 31;
  const int m    = e >> 5;                             // NNBR = 32
  const int j    = (int)nidx[e];

  const float* rn = RN + (size_t)j * RHID;
  const float* rv = RV + (size_t)m * RHID;

  float a0 = 0.f, a1 = 0.f, a2 = 0.f, a3 = 0.f;
#pragma unroll
  for (int q = 0; q < 4; ++q) {
    const int d = lane + (q << 5);
    const float hid = gelu_exact(rn[d] + rv[d]);       // rv includes r_b1
    a0 = fmaf(hid, r_w2[d * 4 + 0], a0);
    a1 = fmaf(hid, r_w2[d * 4 + 1], a1);
    a2 = fmaf(hid, r_w2[d * 4 + 2], a2);
    a3 = fmaf(hid, r_w2[d * 4 + 3], a3);
  }
  a0 = wave_reduce_sum(a0); a1 = wave_reduce_sum(a1);
  a2 = wave_reduce_sum(a2); a3 = wave_reduce_sum(a3);

  if (lane == 0) {
    float4 pi = make_float4(0.f, 0.f, 0.f, 0.f);
    if (nmask[e]) {
      const float l0 = a0 + r_b2[0], l1 = a1 + r_b2[1];
      const float l2 = a2 + r_b2[2], l3 = a3 + r_b2[3];
      const float mx = fmaxf(fmaxf(l0, l1), fmaxf(l2, l3));
      const float e0 = __expf(l0 - mx), e1 = __expf(l1 - mx);
      const float e2 = __expf(l2 - mx), e3 = __expf(l3 - mx);
      const float inv = 1.0f / (e0 + e1 + e2 + e3);
      pi = make_float4(e0 * inv, e1 * inv, e2 * inv, e3 * inv);
    }
    *(float4*)(piTab + (size_t)e * 4) = pi;
  }
}

// ------------------- phase 2b: attention (barrier-free) --------------------
// One block per node m; wave h == head h (Dh=32). After the single preload
// barrier: scores broadcast via xor-reduce, kept lane-resident (lane == n),
// register softmax, coefficients broadcast with __shfl during aggregation.

__global__ __launch_bounds__(256)
void crga_edge_attention(
    const long long* __restrict__ nidx, const unsigned char* __restrict__ nmask,
    const float* __restrict__ piTab,
    const float* __restrict__ Qt,  const float* __restrict__ Qi,
    const float* __restrict__ KPtt, const float* __restrict__ KPit,
    const float* __restrict__ KPti, const float* __restrict__ KPii,
    const float* __restrict__ kt_b, const float* __restrict__ ki_b,
    const float* __restrict__ Ptt, const float* __restrict__ Pit,
    const float* __restrict__ Pti, const float* __restrict__ Pii,
    _Float16* __restrict__ head16_t, _Float16* __restrict__ head16_i)
{
  const int m    = blockIdx.x;
  const int t    = threadIdx.x;
  const int lane = t & 31;

  __shared__ float s_pi[NNBR][4];
  __shared__ int           s_j[NNBR];
  __shared__ unsigned char s_v[NNBR];

  const float qtv = Qt[(size_t)m * DDIM + t];
  const float qiv = Qi[(size_t)m * DDIM + t];
  const float ktb = kt_b[t];
  const float kib = ki_b[t];

  if (t < NNBR) {
    s_j[t] = (int)nidx[(size_t)m * NNBR + t];
    s_v[t] = nmask[(size_t)m * NNBR + t];
  }
  if (t < NNBR * 4) s_pi[t >> 2][t & 3] = piTab[(size_t)m * NNBR * 4 + t];
  __syncthreads();

  // ---- scores: k = pi0*KPtt[j] + pi2*KPit[j] + kt_b (and i-modality) ----
  float sc_t = -3.0e38f, sc_i = -3.0e38f;
  for (int n = 0; n < NNBR; ++n) {
    if (!s_v[n]) continue;                         // uniform per block
    const size_t jo = (size_t)s_j[n] * DDIM + t;
    const float pi0 = s_pi[n][0], pi1 = s_pi[n][1];
    const float pi2 = s_pi[n][2], pi3 = s_pi[n][3];
    float st = qtv * (pi0 * KPtt[jo] + pi2 * KPit[jo] + ktb);
    float si = qiv * (pi1 * KPti[jo] + pi3 * KPii[jo] + kib);
    st = wave_reduce_sum(st);                      // head h == wave h
    si = wave_reduce_sum(si);
    if (lane == n) { sc_t = st * ATT_SCALE; sc_i = si * ATT_SCALE; }
  }

  // ---- register-resident masked softmax over neighbors ----
  const bool vld = s_v[lane] != 0;
  float mx = wave_reduce_max(sc_t);
  float ex = vld ? __expf(sc_t - mx) : 0.f;
  float sm = wave_reduce_sum(ex);
  const float alpha_t = (sm > 0.f) ? (ex / sm) : 0.f;     // nan_to_num

  mx = wave_reduce_max(sc_i);
  ex = vld ? __expf(sc_i - mx) : 0.f;
  sm = wave_reduce_sum(ex);
  const float alpha_i = (sm > 0.f) ? (ex / sm) : 0.f;

  const float my_ctt = alpha_t * s_pi[lane][0];
  const float my_cit = alpha_t * s_pi[lane][2];
  const float my_cti = alpha_i * s_pi[lane][1];
  const float my_cii = alpha_i * s_pi[lane][3];

  // ---- head aggregation: weighted gather-sum of node tables ----
  float acc_t = 0.f, acc_i = 0.f;
  for (int n = 0; n < NNBR; ++n) {
    if (!s_v[n]) continue;
    const size_t jo = (size_t)s_j[n] * DDIM + t;
    const float ctt = __shfl(my_ctt, n, 32);
    const float cit = __shfl(my_cit, n, 32);
    const float cti = __shfl(my_cti, n, 32);
    const float cii = __shfl(my_cii, n, 32);
    acc_t = fmaf(ctt, Ptt[jo], acc_t);
    acc_t = fmaf(cit, Pit[jo], acc_t);
    acc_i = fmaf(cti, Pti[jo], acc_i);
    acc_i = fmaf(cii, Pii[jo], acc_i);
  }
  head16_t[(size_t)m * DDIM + t] = (_Float16)acc_t;
  head16_i[(size_t)m * DDIM + t] = (_Float16)acc_i;
}

// ----------------------- phase 4: residual + layernorm ---------------------

__global__ __launch_bounds__(256)
void crga_residual_ln(const float* __restrict__ h, const float* __restrict__ msg,
                      const float* __restrict__ gamma, const float* __restrict__ beta,
                      float* __restrict__ out)
{
  const int m = blockIdx.x;
  const int t = threadIdx.x;
  __shared__ float s_red[NHEAD];

  const float x = h[(size_t)m * DDIM + t] + msg[(size_t)m * DDIM + t];

  float s = wave_reduce_sum(x);
  if ((t & 31) == 0) s_red[t >> 5] = s;
  __syncthreads();
  float mean = 0.f;
#pragma unroll
  for (int i = 0; i < NHEAD; ++i) mean += s_red[i];
  mean *= (1.0f / DDIM);
  const float d = x - mean;
  __syncthreads();

  float s2 = wave_reduce_sum(d * d);
  if ((t & 31) == 0) s_red[t >> 5] = s2;
  __syncthreads();
  float var = 0.f;
#pragma unroll
  for (int i = 0; i < NHEAD; ++i) var += s_red[i];
  var *= (1.0f / DDIM);

  out[(size_t)m * DDIM + t] = d * rsqrtf(var + 1e-5f) * gamma[t] + beta[t];
}

// ------------------------------ launcher -----------------------------------

extern "C" void kernel_launch(void* const* d_in, const int* in_sizes, int n_in,
                              void* d_out, int out_size, void* d_ws, size_t ws_size,
                              hipStream_t stream)
{
  (void)in_sizes; (void)n_in; (void)out_size; (void)ws_size;

  const float*         h_t   = (const float*)d_in[0];
  const float*         h_i   = (const float*)d_in[1];
  const long long*     nidx  = (const long long*)d_in[2];
  const unsigned char* nmask = (const unsigned char*)d_in[3];
  const float* r_w1 = (const float*)d_in[4];   const float* r_b1 = (const float*)d_in[5];
  const float* r_w2 = (const float*)d_in[6];   const float* r_b2 = (const float*)d_in[7];
  const float* wtt_w = (const float*)d_in[8];  const float* wtt_b = (const float*)d_in[9];
  const float* wti_w = (const float*)d_in[10]; const float* wti_b = (const float*)d_in[11];
  const float* wit_w = (const float*)d_in[12]; const float* wit_b = (const float*)d_in[13];
  const float* wii_w = (const float*)d_in[14]; const float* wii_b = (const float*)d_in[15];
  const float* qt_w = (const float*)d_in[16];  const float* qt_b = (const float*)d_in[17];
  const float* kt_w = (const float*)d_in[18];  const float* kt_b = (const float*)d_in[19];
  const float* qi_w = (const float*)d_in[20];  const float* qi_b = (const float*)d_in[21];
  const float* ki_w = (const float*)d_in[22];  const float* ki_b = (const float*)d_in[23];
  const float* ot_w = (const float*)d_in[24];  const float* ot_b = (const float*)d_in[25];
  const float* oi_w = (const float*)d_in[26];  const float* oi_b = (const float*)d_in[27];
  const float* ln_t_g = (const float*)d_in[28]; const float* ln_t_b = (const float*)d_in[29];
  const float* ln_i_g = (const float*)d_in[30]; const float* ln_i_b = (const float*)d_in[31];

  // ---- workspace layout (fp32 + f16 regions, 16B aligned) ----
  char* ws = (char*)d_ws;
  size_t off = 0;
  auto allocf = [&](size_t n) {
    float* p = (float*)(ws + off); off = (off + n * 4 + 15) & ~(size_t)15; return p;
  };
  auto alloch = [&](size_t n) {
    _Float16* p = (_Float16*)(ws + off); off = (off + n * 2 + 15) & ~(size_t)15; return p;
  };

  const size_t MD = (size_t)MROWS * DDIM;
  const size_t MR = (size_t)MROWS * RHID;

  float* RN   = allocf(MR);
  float* RV   = allocf(MR);
  float* piTab = allocf((size_t)MROWS * NNBR * 4);
  float* Ptt  = allocf(MD);  float* Pit  = allocf(MD);
  float* Pti  = allocf(MD);  float* Pii  = allocf(MD);
  float* KPtt = allocf(MD);  float* KPit = allocf(MD);
  float* KPti = allocf(MD);  float* KPii = allocf(MD);
  float* Qt   = allocf(MD);  float* Qi   = allocf(MD);
  float* msg_t = allocf(MD); float* msg_i = allocf(MD);

  _Float16* h_t16 = alloch(MD);   _Float16* h_i16 = alloch(MD);
  _Float16* Ptt16 = alloch(MD);   _Float16* Pit16 = alloch(MD);
  _Float16* Pti16 = alloch(MD);   _Float16* Pii16 = alloch(MD);
  _Float16* hd16t = alloch(MD);   _Float16* hd16i = alloch(MD);
  _Float16* r1aT = alloch((size_t)RHID * KDIM);
  _Float16* r1bT = alloch((size_t)RHID * KDIM);
  _Float16* r1cT = alloch((size_t)RHID * KDIM);
  _Float16* r1dT = alloch((size_t)RHID * KDIM);
  _Float16* wttT = alloch((size_t)DDIM * KDIM);
  _Float16* wtiT = alloch((size_t)DDIM * KDIM);
  _Float16* witT = alloch((size_t)DDIM * KDIM);
  _Float16* wiiT = alloch((size_t)DDIM * KDIM);
  _Float16* qtT  = alloch((size_t)DDIM * KDIM);
  _Float16* ktT  = alloch((size_t)DDIM * KDIM);
  _Float16* qiT  = alloch((size_t)DDIM * KDIM);
  _Float16* kiT  = alloch((size_t)DDIM * KDIM);
  _Float16* otT  = alloch((size_t)DDIM * KDIM);
  _Float16* oiT  = alloch((size_t)DDIM * KDIM);

  float* out_t = (float*)d_out;
  float* out_i = out_t + MD;

  const dim3 blk(256);

  // ---- stage: f16 activations + transposed f16 weights ----
  const int nMD = (int)MD;
  crga_cvt_f16<<<dim3((nMD + 255) / 256), blk, 0, stream>>>(h_t, h_t16, nMD);
  crga_cvt_f16<<<dim3((nMD + 255) / 256), blk, 0, stream>>>(h_i, h_i16, nMD);

  const dim3 gTr((KDIM * RHID + 255) / 256);
  crga_transpose_f16<<<gTr, blk, 0, stream>>>(r_w1,              r1aT, KDIM, RHID);
  crga_transpose_f16<<<gTr, blk, 0, stream>>>(r_w1 + 256 * RHID, r1bT, KDIM, RHID);
  crga_transpose_f16<<<gTr, blk, 0, stream>>>(r_w1 + 512 * RHID, r1cT, KDIM, RHID);
  crga_transpose_f16<<<gTr, blk, 0, stream>>>(r_w1 + 768 * RHID, r1dT, KDIM, RHID);
  const dim3 gTw((KDIM * DDIM + 255) / 256);
  crga_transpose_f16<<<gTw, blk, 0, stream>>>(wtt_w, wttT, KDIM, DDIM);
  crga_transpose_f16<<<gTw, blk, 0, stream>>>(wti_w, wtiT, KDIM, DDIM);
  crga_transpose_f16<<<gTw, blk, 0, stream>>>(wit_w, witT, KDIM, DDIM);
  crga_transpose_f16<<<gTw, blk, 0, stream>>>(wii_w, wiiT, KDIM, DDIM);
  crga_transpose_f16<<<gTw, blk, 0, stream>>>(qt_w,  qtT,  KDIM, DDIM);
  crga_transpose_f16<<<gTw, blk, 0, stream>>>(kt_w,  ktT,  KDIM, DDIM);
  crga_transpose_f16<<<gTw, blk, 0, stream>>>(qi_w,  qiT,  KDIM, DDIM);
  crga_transpose_f16<<<gTw, blk, 0, stream>>>(ki_w,  kiT,  KDIM, DDIM);
  crga_transpose_f16<<<gTw, blk, 0, stream>>>(ot_w,  otT,  KDIM, DDIM);
  crga_transpose_f16<<<gTw, blk, 0, stream>>>(oi_w,  oiT,  KDIM, DDIM);

  const dim3 gR(MROWS / BLK_M, RHID / BLK_N);   // 64 x 1
  const dim3 gD(MROWS / BLK_M, DDIM / BLK_N);   // 64 x 2

  // ---- phase 1: node-level WMMA GEMMs ----
  crga_gemm_wmma<<<gR, blk, 0, stream>>>(h_t16, r1aT, RN, RHID, nullptr, nullptr, 0);
  crga_gemm_wmma<<<gR, blk, 0, stream>>>(h_i16, r1bT, RN, RHID, nullptr, nullptr, 1);
  crga_gemm_wmma<<<gR, blk, 0, stream>>>(h_t16, r1cT, RV, RHID, r_b1,    nullptr, 0);
  crga_gemm_wmma<<<gR, blk, 0, stream>>>(h_i16, r1dT, RV, RHID, nullptr, nullptr, 1);

  crga_gemm_wmma<<<gD, blk, 0, stream>>>(h_t16, wttT, Ptt, DDIM, wtt_b, Ptt16, 0);
  crga_gemm_wmma<<<gD, blk, 0, stream>>>(h_t16, wtiT, Pti, DDIM, wti_b, Pti16, 0);
  crga_gemm_wmma<<<gD, blk, 0, stream>>>(h_i16, witT, Pit, DDIM, wit_b, Pit16, 0);
  crga_gemm_wmma<<<gD, blk, 0, stream>>>(h_i16, wiiT, Pii, DDIM, wii_b, Pii16, 0);

  crga_gemm_wmma<<<gD, blk, 0, stream>>>(h_t16, qtT, Qt, DDIM, qt_b, nullptr, 0);
  crga_gemm_wmma<<<gD, blk, 0, stream>>>(h_i16, qiT, Qi, DDIM, qi_b, nullptr, 0);

  crga_gemm_wmma<<<gD, blk, 0, stream>>>(Ptt16, ktT, KPtt, DDIM, nullptr, nullptr, 0);
  crga_gemm_wmma<<<gD, blk, 0, stream>>>(Pit16, ktT, KPit, DDIM, nullptr, nullptr, 0);
  crga_gemm_wmma<<<gD, blk, 0, stream>>>(Pti16, kiT, KPti, DDIM, nullptr, nullptr, 0);
  crga_gemm_wmma<<<gD, blk, 0, stream>>>(Pii16, kiT, KPii, DDIM, nullptr, nullptr, 0);

  // ---- phase 2a: router pi per edge (wave-per-edge, no block barriers) ----
  crga_router<<<dim3(MROWS * NNBR / 8), blk, 0, stream>>>(
      nidx, nmask, RN, RV, r_w2, r_b2, piTab);

  // ---- phase 2b: attention + aggregation (L2-resident gathers) ----
  crga_edge_attention<<<dim3(MROWS), blk, 0, stream>>>(
      nidx, nmask, piTab, Qt, Qi,
      KPtt, KPit, KPti, KPii, kt_b, ki_b,
      Ptt, Pit, Pti, Pii, hd16t, hd16i);

  // ---- phase 3: output projections ----
  crga_gemm_wmma<<<gD, blk, 0, stream>>>(hd16t, otT, msg_t, DDIM, ot_b, nullptr, 0);
  crga_gemm_wmma<<<gD, blk, 0, stream>>>(hd16i, oiT, msg_i, DDIM, oi_b, nullptr, 0);

  // ---- phase 4: residual + layernorm -> (h_t_new, h_i_new) ----
  crga_residual_ln<<<dim3(MROWS), blk, 0, stream>>>(h_t, msg_t, ln_t_g, ln_t_b, out_t);
  crga_residual_ln<<<dim3(MROWS), blk, 0, stream>>>(h_i, msg_i, ln_i_g, ln_i_b, out_i);
}
